// WeightOnlyLinear_63754494542163
// MI455X (gfx1250) — matrix operand order repaired and checked by reading, not compiled, not run on hardware
//
#include <hip/hip_runtime.h>

typedef __attribute__((ext_vector_type(16))) _Float16 v16h;
typedef __attribute__((ext_vector_type(8)))  float    v8f;
typedef int v4i_vs __attribute__((vector_size(16)));   // matches builtin param type
typedef __attribute__((address_space(1))) v4i_vs* gptr_b128;
typedef __attribute__((address_space(3))) v4i_vs* lptr_b128;

#define BM  128
#define BN  128
#define BK  32
#define BKP 40   // padded K extent in halves (80 B/row: 16B aligned, staggers banks)

#if __has_builtin(__builtin_amdgcn_global_load_async_to_lds_b128)
#define USE_ASYNC_A 1
#else
#define USE_ASYNC_A 0
#endif

__device__ __forceinline__ void wait_async_lds() {
#if USE_ASYNC_A
#if __has_builtin(__builtin_amdgcn_s_wait_asynccnt)
    __builtin_amdgcn_s_wait_asynccnt(0);
#else
    asm volatile("s_wait_asynccnt 0x0" ::: "memory");
#endif
#endif
}

__global__ __launch_bounds__(256)
void w8a16_wmma_gemm(const _Float16* __restrict__ x,
                     const int*      __restrict__ qw,
                     const _Float16* __restrict__ scale,
                     const _Float16* __restrict__ bias,
                     _Float16*       __restrict__ out,
                     int M, int K, int N)
{
    __shared__ _Float16 ldsA[2][BM * BKP];   // [m][k] fp16
    __shared__ _Float16 ldsB[2][BN * BKP];   // [n][k] fp16 (transposed during load)

    const int t    = threadIdx.x;
    const int lane = t & 31;
    const int wave = t >> 5;
    const int wm   = wave & 3;        // rows wm*32 .. +32
    const int wn   = wave >> 2;       // cols wn*64 .. +64

    // M-tiles on blockIdx.x (fastest-varying): the concurrent block window spans
    // all 64 M-tiles x few N-panels, so weight panels + all A row-panels stay in
    // the 192MB L2 and HBM streams each tensor ~once (~0.42 GB -> ~18us).
    const int m0 = blockIdx.x * BM;
    const int n0 = blockIdx.y * BN;

    // cooperative loader indices
    const int la_m  = t >> 1;               // 0..127 : A row
    const int la_kh = (t & 1) << 4;         // 0 or 16: A k-half
    const int lb_n  = (t & 31) << 2;        // 0..124 step 4 : B n-group
    const int lb_k  = (t >> 5) << 2;        // 0..28  step 4 : B k-group

    // WMMA fragment lane geometry (per ISA 16-bit layouts)
    const int lsub  = lane & 15;
    const int abase = (lane < 16) ? 0 : 8;
    const int bbase = (lane < 16) ? 0 : 8;

    const _Float16* gA = x  + (size_t)(m0 + la_m) * K + la_kh;
    const int*      gB = qw + (size_t)lb_k * N + n0 + lb_n;

    v8f acc[2][4];
    #pragma unroll
    for (int mi = 0; mi < 2; ++mi)
        #pragma unroll
        for (int ni = 0; ni < 4; ++ni)
            acc[mi][ni] = (v8f){0.f, 0.f, 0.f, 0.f, 0.f, 0.f, 0.f, 0.f};

    // stage registers (fetch fills them early; commit consumes them late)
    uint4 areg0, areg1;
    int4  breg[4];

    auto fetch = [&](int buf, int k0) {
#if USE_ASYNC_A
        // DMA the fp16 A tile straight into LDS; no VGPR round-trip.
        __builtin_amdgcn_global_load_async_to_lds_b128(
            (gptr_b128)(gA + k0),
            (lptr_b128)&ldsA[buf][la_m * BKP + la_kh],
            0, 0);
        __builtin_amdgcn_global_load_async_to_lds_b128(
            (gptr_b128)(gA + k0),
            (lptr_b128)&ldsA[buf][la_m * BKP + la_kh],
            16, 0);
#else
        const _Float16* ga = gA + k0;
        areg0 = *(const uint4*)(ga);
        areg1 = *(const uint4*)(ga + 8);
#endif
        const int* gb = gB + (size_t)k0 * N;
        #pragma unroll
        for (int kk = 0; kk < 4; ++kk)
            breg[kk] = *(const int4*)(gb + (size_t)kk * N);
    };

    auto commit = [&](int buf) {
#if !USE_ASYNC_A
        _Float16* sa = &ldsA[buf][la_m * BKP + la_kh];
        *(uint4*)(sa)     = areg0;
        *(uint4*)(sa + 8) = areg1;
#endif
        #pragma unroll
        for (int nn = 0; nn < 4; ++nn) {
            union { _Float16 h[4]; uint2 u; } col;
            col.h[0] = (_Float16)((&breg[0].x)[nn]);
            col.h[1] = (_Float16)((&breg[1].x)[nn]);
            col.h[2] = (_Float16)((&breg[2].x)[nn]);
            col.h[3] = (_Float16)((&breg[3].x)[nn]);
            *(uint2*)&ldsB[buf][(lb_n + nn) * BKP + lb_k] = col.u;
        }
    };

    // prologue
    fetch(0, 0);
    commit(0);
    wait_async_lds();
    __syncthreads();

    const int ksteps = K / BK;
    for (int ks = 0; ks < ksteps; ++ks) {
        const int cur = ks & 1;
        const bool hasNext = (ks + 1) < ksteps;
        if (hasNext)
            fetch(cur ^ 1, (ks + 1) * BK);   // issue global loads before compute

        // A fragments: halves [0..7] = K abase..+7, [8..15] = K abase+16..+23
        v16h afrag[2];
        #pragma unroll
        for (int mi = 0; mi < 2; ++mi) {
            const _Float16* pa = &ldsA[cur][(wm * 32 + mi * 16 + lsub) * BKP + abase];
            ((uint4*)&afrag[mi])[0] = *(const uint4*)(pa);
            ((uint4*)&afrag[mi])[1] = *(const uint4*)(pa + 16);
        }
        // B fragments: halves [0..7] = K bbase..+7, [8..15] = K bbase+16..+23
        v16h bfrag[4];
        #pragma unroll
        for (int ni = 0; ni < 4; ++ni) {
            const _Float16* pb = &ldsB[cur][(wn * 64 + ni * 16 + lsub) * BKP + bbase];
            ((uint4*)&bfrag[ni])[0] = *(const uint4*)(pb);
            ((uint4*)&bfrag[ni])[1] = *(const uint4*)(pb + 16);
        }

        #pragma unroll
        for (int mi = 0; mi < 2; ++mi)
            #pragma unroll
            for (int ni = 0; ni < 4; ++ni)
                acc[mi][ni] = __builtin_amdgcn_wmma_f32_16x16x32_f16(
                    false, afrag[mi], false, bfrag[ni],
                    (short)0, acc[mi][ni], false, false);

        if (hasNext)
            commit(cur ^ 1);                 // convert + DS-store after compute

        wait_async_lds();
        __syncthreads();
    }

    // epilogue: f32 scale, then fp16 bias add (matches reference)
    // C/D layout: lane holds (m = v + 8*(lane>=16), n = lane%16) for VGPR v
    const int mrow = (lane < 16) ? 0 : 8;
    #pragma unroll
    for (int ni = 0; ni < 4; ++ni) {
        const int gn = n0 + wn * 64 + ni * 16 + lsub;
        const float    sc = (float)scale[gn];
        const _Float16 bs = bias[gn];
        #pragma unroll
        for (int mi = 0; mi < 2; ++mi) {
            const int rowbase = m0 + wm * 32 + mi * 16 + mrow;
            #pragma unroll
            for (int v = 0; v < 8; ++v) {
                _Float16 r = (_Float16)(acc[mi][ni][v] * sc);
                r = (_Float16)(r + bs);
                out[(size_t)(rowbase + v) * N + gn] = r;
            }
        }
    }
}

extern "C" void kernel_launch(void* const* d_in, const int* in_sizes, int n_in,
                              void* d_out, int out_size, void* d_ws, size_t ws_size,
                              hipStream_t stream) {
    const _Float16* x     = (const _Float16*)d_in[0];
    const int*      qw    = (const int*)d_in[1];
    const _Float16* scale = (const _Float16*)d_in[2];
    const _Float16* bias  = (const _Float16*)d_in[3];
    _Float16*       out   = (_Float16*)d_out;

    const int N = in_sizes[2];            // 11008
    const int K = in_sizes[1] / N;        // 4096
    const int M = in_sizes[0] / K;        // 8192

    dim3 grid(M / BM, N / BN);            // 64 x 86 blocks, M fastest
    w8a16_wmma_gemm<<<grid, 256, 0, stream>>>(x, qw, scale, bias, out, M, K, N);
}
